// PyTorchConv2dAutograd_8083128451540
// MI455X (gfx1250) — matrix operand order repaired and checked by reading, not compiled, not run on hardware
//
#include <hip/hip_runtime.h>
#include <hip/hip_bf16.h>

// Implicit-GEMM conv2d (valid cross-correlation) for MI455X / gfx1250.
//   out[n,f,oy,ox] = sum_{c,p,q} x[n,c,oy+p,ox+q] * w[f,c,p,q] + b[f]
// GEMM: D[M=128][N] = W[M][K=1600] * X[K][N], K = 25 taps (p,q) x 2 c-blocks.
// N-tile = 4 output rows x 32 cols (regular 3D tensor slab); W and X tiles are
// fetched by the Tensor Data Mover (tensor_load_to_lds, TENSORcnt) issued by
// wave 0, while 8 waves run bf16 WMMA (4 accumulators each, shared A frag)
// out of double-buffered LDS.
//
// prep_w : w fp32 -> Wr bf16 [25*2][128][32]      (GEMM-ready, contiguous)
// prep_x : x NCHW fp32 -> Xr bf16 NHWC (32,64,64,64)
// Workspace: Wr at d_ws+0 (400KB), Xr at d_ws+512KB (16MB).

typedef __bf16 bf16_t;
typedef __attribute__((ext_vector_type(16))) __bf16 v16bf;
typedef __attribute__((ext_vector_type(8)))  __bf16 v8bf;
typedef __attribute__((ext_vector_type(8)))  float  v8f;
typedef __attribute__((ext_vector_type(4)))  unsigned int u32x4;
typedef __attribute__((ext_vector_type(8)))  int i32x8;
typedef __attribute__((ext_vector_type(4)))  int i32x4;

#define TM     64     // block tile M (filters)
#define TN     128    // block tile N = 4 rows x 32 cols
#define TK     32     // channels per WMMA k-slice
#define KSTEPS 50     // 25 taps * 2 channel blocks

static __device__ __forceinline__ bf16_t f2bf(float f) {
    unsigned u = __builtin_bit_cast(unsigned, f);
    u += 0x7FFFu + ((u >> 16) & 1u);
    unsigned short s = (unsigned short)(u >> 16);
    return __builtin_bit_cast(bf16_t, s);
}

// ---------------------------------------------------------------------------
// Prep 1: weights fp32 (128,64,5,5) -> bf16 [pq][cblk][f][cj]
// ---------------------------------------------------------------------------
__global__ __launch_bounds__(256)
void prep_w_kernel(const float* __restrict__ w, bf16_t* __restrict__ Wr) {
    const int idx  = blockIdx.x * 256 + threadIdx.x;   // 0 .. 204799
    const int cj   = idx & 31;
    const int f    = (idx >> 5) & 127;
    const int cblk = (idx >> 12) & 1;
    const int pq   = idx >> 13;                        // 0 .. 24
    const float v = w[(size_t)f * 1600 + (size_t)(cblk * 32 + cj) * 25 + pq];
    Wr[idx] = f2bf(v);
}

// ---------------------------------------------------------------------------
// Prep 2: x NCHW fp32 -> NHWC bf16 via LDS-tiled transpose (block per (n,h)).
// ---------------------------------------------------------------------------
__global__ __launch_bounds__(256)
void prep_x_kernel(const float* __restrict__ x, bf16_t* __restrict__ Xr) {
    __shared__ float Xs[64][68];
    const int n = blockIdx.x >> 6;
    const int h = blockIdx.x & 63;
    const int t = threadIdx.x;
    {
        const int c  = t >> 2;
        const int w0 = (t & 3) * 16;
        const float4* src =
            (const float4*)(x + (((size_t)(n * 64 + c) * 64 + h) * 64 + w0));
        float4 a = src[0], b = src[1], c2 = src[2], d = src[3];
        float4* dst = (float4*)&Xs[c][w0];
        dst[0] = a; dst[1] = b; dst[2] = c2; dst[3] = d;
    }
    __syncthreads();
    {
        const int wi = t >> 2;
        const int c0 = (t & 3) * 16;
        unsigned pk[8];
#pragma unroll
        for (int j = 0; j < 8; ++j) {
            unsigned lo = (unsigned)__builtin_bit_cast(unsigned short, f2bf(Xs[c0 + 2 * j][wi]));
            unsigned hi = (unsigned)__builtin_bit_cast(unsigned short, f2bf(Xs[c0 + 2 * j + 1][wi]));
            pk[j] = lo | (hi << 16);
        }
        bf16_t* dstb = Xr + (((size_t)(n * 64 + h) * 64 + wi) * 64 + c0);
        ((uint4*)dstb)[0] = make_uint4(pk[0], pk[1], pk[2], pk[3]);
        ((uint4*)dstb)[1] = make_uint4(pk[4], pk[5], pk[6], pk[7]);
    }
}

// ---------------------------------------------------------------------------
// Main: bf16 WMMA implicit GEMM, tiles loaded by the Tensor Data Mover.
// Block = 256 thr = 8 waves (4 along M x 2 along N), tile 64(M) x 128(N).
// ---------------------------------------------------------------------------
__global__ __launch_bounds__(256)
void conv2d_wmma_kernel(const bf16_t* __restrict__ Xr,
                        const bf16_t* __restrict__ Wr,
                        const float* __restrict__ bias,
                        float* __restrict__ out) {
    __shared__ __align__(64) bf16_t Wl[2][TM][TK];     // 4KB * 2
    __shared__ __align__(64) bf16_t Xl[2][TN][TK];     // 8KB * 2

    const int tid   = threadIdx.x;
    const int lane  = tid & 31;
    const int wv    = tid >> 5;
    const int waveM = wv & 3;
    const int waveN = wv >> 2;

    // block -> (n, oy0, ox0) x bm    grid.x = 32 * 15 * 2 = 960
    const int bx    = blockIdx.x;
    const int bm    = blockIdx.y;                      // 0..1
    const int ox0   = (bx & 1) * 32;                   // 0, 32
    const int rest  = bx >> 1;
    const int n_img = rest / 15;
    const int oy0   = (rest - n_img * 15) * 4;

    const unsigned long long wr_base = (unsigned long long)(size_t)Wr;
    const unsigned long long xr_base = (unsigned long long)(size_t)Xr;

    // ---- TDM descriptor issue (wave-uniform scalars only) -------------------
    auto issue_tiles = [&](int buf, int kt) {
        const int pq   = kt >> 1;
        const int cblk = kt & 1;
        const int p    = pq / 5;
        const int q    = pq - p * 5;
        const i32x8 gz = {0, 0, 0, 0, 0, 0, 0, 0};     // 6th-operand group: zeros

        // W tile: 2D  [64 f][32 c], row stride 32 elems
        {
            const unsigned lds = (unsigned)(size_t)&Wl[buf][0][0];
            const unsigned long long ga =
                wr_base + (unsigned long long)((kt * 128 + bm * TM) * TK) * 2ull;
            u32x4 g0 = {1u,                                   // count=1
                        lds,                                  // lds_addr
                        (unsigned)ga,                         // global_addr[31:0]
                        ((unsigned)(ga >> 32) & 0x1FFFFFFu) | 0x80000000u}; // type=2
            i32x8 g1 = {0x10000,        // data_size = 2B
                        32 << 16,       // tensor_dim0 = 32
                        8192 << 16,     // tensor_dim1 = 8192 (rows, never OOB)
                        32 << 16,       // tile_dim0 = 32
                        64,             // tile_dim1 = 64, tile_dim2 = 0 (2D)
                        32,             // tensor_dim0_stride = 32
                        0, 0};
            i32x4 g2 = {1, 1, 0, 0};
            i32x4 g3 = {0, 0, 0, 0};
            __builtin_amdgcn_tensor_load_to_lds(g0, g1, g2, g3, gz, 0);
        }
        // X tile: 3D  [4 rows][32 px][32 c], strides [4096][64][1] over NHWC
        {
            const unsigned lds = (unsigned)(size_t)&Xl[buf][0][0];
            const int elem = ((n_img * 64 + oy0 + p) * 64 + (ox0 + q)) * 64 + cblk * TK;
            const unsigned long long ga = xr_base + (unsigned long long)elem * 2ull;
            const int td1 = 64 - ox0 - q;  // px remaining in row; beyond -> zeros
            u32x4 g0 = {1u,
                        lds,
                        (unsigned)ga,
                        ((unsigned)(ga >> 32) & 0x1FFFFFFu) | 0x80000000u};
            i32x8 g1 = {0x10000,        // data_size = 2B
                        32 << 16,       // tensor_dim0 = 32 (c window)
                        td1 << 16,      // tensor_dim1 = remaining px in row
                        32 << 16,       // tile_dim0 = 32
                        32 | (4 << 16), // tile_dim1 = 32 px, tile_dim2 = 4 rows
                        64,             // tensor_dim0_stride = 64 (px step)
                        4096 << 16,     // tensor_dim1_stride = 4096 (row step), lo16
                        0};             // stride1 hi
            i32x4 g2 = {4, 1, 0, 0};    // tensor_dim2 = 4 (rows in tile, in-bounds)
            i32x4 g3 = {0, 0, 0, 0};
            __builtin_amdgcn_tensor_load_to_lds(g0, g1, g2, g3, gz, 0);
        }
    };

    // ---- per-lane fragment coordinates (contiguous k = 16*(lane/16) + s for A and B) ----------
    const int col = lane & 15;
    const int am  = waveM * 16 + col;             // A row (filter)
    const int ak0 = (lane < 16) ? 0 : 16;         // A: 16 contiguous k
    const int bp  = waveN * 64 + col;             // B column base (position)
    const int bk0 = (lane < 16) ? 0 : 16;         // B: 16 contiguous k

    v8f acc[4] = {v8f{}, v8f{}, v8f{}, v8f{}};

    if (wv == 0) issue_tiles(0, 0);
    __builtin_amdgcn_s_wait_tensorcnt(0);
    __syncthreads();

#pragma unroll 2
    for (int kt = 0; kt < KSTEPS; ++kt) {
        const int cur = kt & 1;
        if (wv == 0 && kt + 1 < KSTEPS) issue_tiles(1 - cur, kt + 1);

        v8bf alo = *(const v8bf*)&Wl[cur][am][ak0];
        v8bf ahi = *(const v8bf*)&Wl[cur][am][ak0 + 8];
        v16bf a = __builtin_shufflevector(alo, ahi,
                       0, 1, 2, 3, 4, 5, 6, 7, 8, 9, 10, 11, 12, 13, 14, 15);
#pragma unroll
        for (int t = 0; t < 4; ++t) {
            v16bf b = *(const v16bf*)&Xl[cur][bp + t * 16][bk0];
            acc[t] = __builtin_amdgcn_wmma_f32_16x16x32_bf16(
                         false, a, false, b, (short)0, acc[t], false, false);
        }

        __builtin_amdgcn_s_wait_tensorcnt(0);   // no-op for non-issuing waves
        __syncthreads();
    }

    // ---- epilogue: C/D layout (VGPR r -> M = r + 8*(lane>=16), N = lane&15)
    // acc[t] covers tile position i = waveN*64 + t*16 + col;
    // tile is 4 rows x 32 cols: oy = oy0 + (i>>5), ox = ox0 + (i&31).
    const int fBase = bm * TM + waveM * 16 + ((lane >> 4) << 3);
    const size_t nb = (size_t)n_img * 128;
#pragma unroll
    for (int t = 0; t < 4; ++t) {
        const int i  = waveN * 64 + t * 16 + col;
        const int oy = oy0 + (i >> 5);
        const int ox = ox0 + (i & 31);
        if (ox < 60) {
#pragma unroll
            for (int r = 0; r < 8; ++r) {
                const int f = fBase + r;
                out[((nb + f) * 60 + oy) * 60 + ox] = acc[t][r] + bias[f];
            }
        }
    }
}

// ---------------------------------------------------------------------------
extern "C" void kernel_launch(void* const* d_in, const int* in_sizes, int n_in,
                              void* d_out, int out_size, void* d_ws, size_t ws_size,
                              hipStream_t stream) {
    const float* x    = (const float*)d_in[0];  // (32,64,64,64) NCHW
    const float* w    = (const float*)d_in[1];  // (128,64,5,5)
    const float* bias = (const float*)d_in[2];  // (128,)
    float* out        = (float*)d_out;          // (32,128,60,60)

    bf16_t* Wr = (bf16_t*)d_ws;                             // 409,600 B
    bf16_t* Xr = (bf16_t*)((char*)d_ws + (512 * 1024));     // 16,777,216 B

    prep_w_kernel<<<dim3(800), dim3(256), 0, stream>>>(w, Wr);
    prep_x_kernel<<<dim3(2048), dim3(256), 0, stream>>>(x, Xr);

    dim3 grid(32 * 15 * 2, 2);                              // (960, 2)
    conv2d_wmma_kernel<<<grid, dim3(256), 0, stream>>>(Xr, Wr, bias, out);
}
